// DynamicRelationExtraction_79525614453050
// MI455X (gfx1250) — matrix operand order repaired and checked by reading, compiled, not run on hardware
//
#include <hip/hip_runtime.h>
#include <hip/hip_bf16.h>
#include <math.h>
#include <stdint.h>

typedef __attribute__((ext_vector_type(16))) __bf16 v16bf;
typedef __attribute__((ext_vector_type(8)))  __bf16 v8bf;
typedef __attribute__((ext_vector_type(8)))  float  v8f;
typedef int gv4i __attribute__((vector_size(16)));   // gcc-style int4 (builtin param type)

#define N_NODES 8192
#define K_FREQ  256
#define KX      512      // concatenated [Ac|As] width
#define C_KEEP  32
#define EPS_F   1e-10f

#define AS1 __attribute__((address_space(1)))
#define AS3 __attribute__((address_space(3)))

// ---------------------------------------------------------------------------
// async global->LDS copy (ASYNCcnt-tracked), builtin if present else inline asm
// ---------------------------------------------------------------------------
#if __has_builtin(__builtin_amdgcn_global_load_async_to_lds_b128)
#define ASYNC_B128(g, l)                                                      \
    __builtin_amdgcn_global_load_async_to_lds_b128(                           \
        (AS1 gv4i*)(uintptr_t)(g), (AS3 gv4i*)(uintptr_t)(l), 0, 0)
#else
#define ASYNC_B128(g, l)                                                      \
    asm volatile("global_load_async_to_lds_b128 %0, %1, off"                  \
                 :: "v"((unsigned)(uintptr_t)(l)),                            \
                    "v"((const void*)(g)) : "memory")
#endif

#if __has_builtin(__builtin_amdgcn_s_wait_asynccnt)
#define WAIT_ASYNC(n) __builtin_amdgcn_s_wait_asynccnt(n)
#else
#define WAIT_ASYNC(n) asm volatile("s_wait_asynccnt %0" :: "i"(n) : "memory")
#endif

// XOR swizzle of the four 16B chunks within a 64B LDS row (conflict spreading)
__device__ __forceinline__ int swz(int r) { return (r & 3) ^ ((r >> 2) & 3); }

// ---------------------------------------------------------------------------
// Kernel 1: top-32 mask, Ac/As, bf16 hi/lo split, sq[n]
// ---------------------------------------------------------------------------
__global__ __launch_bounds__(256)
void prep_kernel(const float* __restrict__ P, const float* __restrict__ A,
                 __bf16* __restrict__ Xhi, __bf16* __restrict__ Xlo,
                 float* __restrict__ sq)
{
    __shared__ float lds[K_FREQ];
    const int row = blockIdx.x;
    const int t   = threadIdx.x;
    const size_t base = (size_t)row * K_FREQ;

    float a = A[base + t];
    lds[t] = a;
    __syncthreads();

    // exact rank (ties broken by smaller index, matching lax.top_k)
    int rank = 0;
    #pragma unroll 8
    for (int j = 0; j < K_FREQ; ++j) {
        float o = lds[j];
        rank += (o > a) || (o == a && j < t);
    }
    float am = (rank < C_KEEP) ? a : 0.0f;

    float p = P[base + t];
    float sp, cp;
    sincosf(p, &sp, &cp);
    float ac = am * cp;
    float as = am * sp;

    const size_t xb = (size_t)row * KX;
    __bf16 h0 = (__bf16)ac;
    __bf16 h1 = (__bf16)as;
    Xhi[xb + t]          = h0;
    Xhi[xb + K_FREQ + t] = h1;
    Xlo[xb + t]          = (__bf16)(ac - (float)h0);
    Xlo[xb + K_FREQ + t] = (__bf16)(as - (float)h1);

    __syncthreads();
    lds[t] = am * am;
    __syncthreads();
    for (int s = 128; s > 0; s >>= 1) {
        if (t < s) lds[t] += lds[t + s];
        __syncthreads();
    }
    if (t == 0) sq[row] = lds[0];
}

// ---------------------------------------------------------------------------
// Kernel 2: R = X X^T / sqrt(sq_i*sq_j+eps), split-bf16 WMMA
// 128x128 tile / WG, 8 waves @ 64x32; async double-buffered LDS staging
// smem[buf][mat][128*32], mat: 0=Ahi 1=Alo 2=Bhi 3=Blo  (exactly 64 KB)
// ---------------------------------------------------------------------------
__global__ __launch_bounds__(256)
void gemm_kernel(const __bf16* __restrict__ Xhi, const __bf16* __restrict__ Xlo,
                 const float* __restrict__ sq, float* __restrict__ R)
{
    __shared__ __bf16 smem[2][4][128 * 32];

    const int tile_i = blockIdx.y * 128;
    const int tile_j = blockIdx.x * 128;
    const int tid  = threadIdx.x;
    const int wave = tid >> 5;
    const int lane = tid & 31;
    const int wrow = (wave >> 2) * 64;   // 0 or 64
    const int wcol = (wave & 3) * 32;    // 0,32,64,96
    const int lm   = lane & 15;
    const int hi16 = lane >> 4;          // 0 or 1

    v8f acc[4][2];
    #pragma unroll
    for (int i = 0; i < 4; ++i)
        #pragma unroll
        for (int j = 0; j < 2; ++j)
            acc[i][j] = (v8f){0.f,0.f,0.f,0.f,0.f,0.f,0.f,0.f};

    union V16 { v16bf v; v8bf h[2]; };

    // issue one 128x32 chunk (4 matrices) of async loads: 8 b128 per thread
    auto issue = [&](int buf, int kb) {
        #pragma unroll
        for (int h = 0; h < 2; ++h) {
            int c = tid + h * 256;           // 0..511 chunk id per matrix
            int r = c >> 2;                  // row in tile
            int k = c & 3;                   // 16B chunk within 32-bf16 row
            int loff = r * 32 + (k ^ swz(r)) * 8;
            size_t ga = (size_t)(tile_i + r) * KX + kb + k * 8;
            size_t gb = (size_t)(tile_j + r) * KX + kb + k * 8;
            ASYNC_B128(Xhi + ga, &smem[buf][0][loff]);
            ASYNC_B128(Xlo + ga, &smem[buf][1][loff]);
            ASYNC_B128(Xhi + gb, &smem[buf][2][loff]);
            ASYNC_B128(Xlo + gb, &smem[buf][3][loff]);
        }
    };

    issue(0, 0);

    // A frag chunks: lane<16 -> K{0-7,16-23} = chunks {0,2}; lane>=16 -> {1,3}
    // B frag chunks: lane<16 -> K{0-7,16-23} = chunks {0,2}; lane>=16 -> {1,3}
    const int ca0 = hi16 ? 1 : 0, ca1 = hi16 ? 3 : 2;
    const int cb0 = hi16 ? 1 : 0, cb1 = hi16 ? 3 : 2;

    for (int it = 0; it < KX / 32; ++it) {
        const int cur = it & 1;
        const int kb  = it * 32;
        if (it + 1 < KX / 32) {
            issue(cur ^ 1, kb + 32);
            WAIT_ASYNC(8);                 // 8 newest = next buf; cur buf done
        } else {
            WAIT_ASYNC(0);
        }
        __syncthreads();                   // everyone's cur-buffer data in LDS

        v16bf ahi[4], alo[4], bhi[2], blo[2];
        #pragma unroll
        for (int mf = 0; mf < 4; ++mf) {
            int m = wrow + mf * 16 + lm;
            int sw = swz(m);
            const __bf16* pH = &smem[cur][0][m * 32];
            const __bf16* pL = &smem[cur][1][m * 32];
            V16 th, tl;
            th.h[0] = *(const v8bf*)(pH + ((ca0 ^ sw) << 3));
            th.h[1] = *(const v8bf*)(pH + ((ca1 ^ sw) << 3));
            tl.h[0] = *(const v8bf*)(pL + ((ca0 ^ sw) << 3));
            tl.h[1] = *(const v8bf*)(pL + ((ca1 ^ sw) << 3));
            ahi[mf] = th.v; alo[mf] = tl.v;
        }
        #pragma unroll
        for (int nf = 0; nf < 2; ++nf) {
            int n = wcol + nf * 16 + lm;
            int sw = swz(n);
            const __bf16* pH = &smem[cur][2][n * 32];
            const __bf16* pL = &smem[cur][3][n * 32];
            V16 th, tl;
            th.h[0] = *(const v8bf*)(pH + ((cb0 ^ sw) << 3));
            th.h[1] = *(const v8bf*)(pH + ((cb1 ^ sw) << 3));
            tl.h[0] = *(const v8bf*)(pL + ((cb0 ^ sw) << 3));
            tl.h[1] = *(const v8bf*)(pL + ((cb1 ^ sw) << 3));
            bhi[nf] = th.v; blo[nf] = tl.v;
        }

        // split-bf16: hi*hi + hi*lo + lo*hi, fp32 accumulate
        #pragma unroll
        for (int mf = 0; mf < 4; ++mf)
            #pragma unroll
            for (int nf = 0; nf < 2; ++nf) {
                acc[mf][nf] = __builtin_amdgcn_wmma_f32_16x16x32_bf16(
                    false, ahi[mf], false, bhi[nf], (short)0, acc[mf][nf], false, false);
                acc[mf][nf] = __builtin_amdgcn_wmma_f32_16x16x32_bf16(
                    false, ahi[mf], false, blo[nf], (short)0, acc[mf][nf], false, false);
                acc[mf][nf] = __builtin_amdgcn_wmma_f32_16x16x32_bf16(
                    false, alo[mf], false, bhi[nf], (short)0, acc[mf][nf], false, false);
            }

        __syncthreads();                   // everyone done reading cur buffer
    }

    // epilogue: C/D layout -> VGPR r, lane l : M = r + 8*(l>=16), N = l&15
    #pragma unroll
    for (int nf = 0; nf < 2; ++nf) {
        int colg = tile_j + wcol + nf * 16 + lm;
        float sqc = sq[colg];
        #pragma unroll
        for (int mf = 0; mf < 4; ++mf) {
            #pragma unroll
            for (int vr = 0; vr < 8; ++vr) {
                int rowg = tile_i + wrow + mf * 16 + vr + hi16 * 8;
                float d = sqrtf(sq[rowg] * sqc + EPS_F);
                R[(size_t)rowg * N_NODES + colg] = acc[mf][nf][vr] / d;
            }
        }
    }
}

// ---------------------------------------------------------------------------
// Kernel 3: per-row sum, positive/negative sums & counts
// ---------------------------------------------------------------------------
__global__ __launch_bounds__(256)
void rowstats_kernel(const float* __restrict__ R, float* __restrict__ rowsum,
                     float* __restrict__ psum, float* __restrict__ pcnt,
                     float* __restrict__ nsum, float* __restrict__ ncnt)
{
    __shared__ float red[256];
    const int row = blockIdx.x;
    const int t   = threadIdx.x;
    const float* rp = R + (size_t)row * N_NODES;

    float s = 0.f, ps = 0.f, pc = 0.f, ns = 0.f, nc = 0.f;
    for (int j = t * 4; j < N_NODES; j += 256 * 4) {
        float4 v = *(const float4*)(rp + j);
        float xs[4] = {v.x, v.y, v.z, v.w};
        #pragma unroll
        for (int u = 0; u < 4; ++u) {
            float x = xs[u];
            s += x;
            if (x > 0.f)      { ps += x; pc += 1.f; }
            else if (x < 0.f) { ns += x; nc += 1.f; }
        }
    }
#define BLK_REDUCE(val, dst)                                            \
    red[t] = (val); __syncthreads();                                    \
    for (int ss = 128; ss > 0; ss >>= 1) {                              \
        if (t < ss) red[t] += red[t + ss];                              \
        __syncthreads();                                                \
    }                                                                   \
    if (t == 0) (dst)[row] = red[0];                                    \
    __syncthreads();

    BLK_REDUCE(s,  rowsum)
    BLK_REDUCE(ps, psum)
    BLK_REDUCE(pc, pcnt)
    BLK_REDUCE(ns, nsum)
    BLK_REDUCE(nc, ncnt)
#undef BLK_REDUCE
}

// ---------------------------------------------------------------------------
// Kernel 4: tau = mean of strictly-positive R' entries (single block)
// ---------------------------------------------------------------------------
__global__ __launch_bounds__(256)
void tau_kernel(const float* __restrict__ rowsum, const float* __restrict__ psum,
                const float* __restrict__ pcnt, const float* __restrict__ nsum,
                const float* __restrict__ ncnt, float* __restrict__ tau)
{
    __shared__ float r1[256], r2[256];
    const int t = threadIdx.x;
    float ms = 0.f, mc = 0.f;
    for (int r = t; r < N_NODES; r += 256) {
        float sden = rowsum[r] + EPS_F;
        if (sden > 0.f)      { ms += psum[r] / sden; mc += pcnt[r]; }
        else if (sden < 0.f) { ms += nsum[r] / sden; mc += ncnt[r]; }
    }
    r1[t] = ms; r2[t] = mc;
    __syncthreads();
    for (int s = 128; s > 0; s >>= 1) {
        if (t < s) { r1[t] += r1[t + s]; r2[t] += r2[t + s]; }
        __syncthreads();
    }
    if (t == 0) {
        float mean = r1[0] / fmaxf(r2[0], 1.0f);
        tau[0] = (mean > 0.f) ? mean : 1.0f;   // TAU_FACTOR = 1.0
    }
}

// ---------------------------------------------------------------------------
// Kernel 5: R_hat = where(R/rowsum >= tau, R/rowsum, 0), in place, float4
// ---------------------------------------------------------------------------
__global__ __launch_bounds__(256)
void finalize_kernel(float* __restrict__ R, const float* __restrict__ rowsum,
                     const float* __restrict__ tau_p)
{
    const float tau = tau_p[0];
    const size_t total4 = (size_t)N_NODES * N_NODES / 4;
    size_t idx = (size_t)blockIdx.x * blockDim.x + threadIdx.x;
    const size_t stride = (size_t)gridDim.x * blockDim.x;
    for (; idx < total4; idx += stride) {
        int row = (int)((idx * 4) >> 13);    // 8192 cols per row
        float inv = 1.0f / (rowsum[row] + EPS_F);
        float4 v = ((float4*)R)[idx];
        float x;
        x = v.x * inv; v.x = (x >= tau) ? x : 0.f;
        x = v.y * inv; v.y = (x >= tau) ? x : 0.f;
        x = v.z * inv; v.z = (x >= tau) ? x : 0.f;
        x = v.w * inv; v.w = (x >= tau) ? x : 0.f;
        ((float4*)R)[idx] = v;
    }
}

// ---------------------------------------------------------------------------
extern "C" void kernel_launch(void* const* d_in, const int* in_sizes, int n_in,
                              void* d_out, int out_size, void* d_ws, size_t ws_size,
                              hipStream_t stream)
{
    const float* P = (const float*)d_in[0];   // P_ref [N,K]
    const float* A = (const float*)d_in[1];   // A_ref [N,K]
    float* R = (float*)d_out;                 // [N,N] staged + final output

    char* ws = (char*)d_ws;
    size_t off = 0;
    __bf16* Xhi = (__bf16*)(ws + off); off += (size_t)N_NODES * KX * sizeof(__bf16);
    __bf16* Xlo = (__bf16*)(ws + off); off += (size_t)N_NODES * KX * sizeof(__bf16);
    float* sq     = (float*)(ws + off); off += N_NODES * sizeof(float);
    float* rowsum = (float*)(ws + off); off += N_NODES * sizeof(float);
    float* psum   = (float*)(ws + off); off += N_NODES * sizeof(float);
    float* pcnt   = (float*)(ws + off); off += N_NODES * sizeof(float);
    float* nsum   = (float*)(ws + off); off += N_NODES * sizeof(float);
    float* ncnt   = (float*)(ws + off); off += N_NODES * sizeof(float);
    float* tau    = (float*)(ws + off); off += sizeof(float);

    prep_kernel<<<N_NODES, 256, 0, stream>>>(P, A, Xhi, Xlo, sq);

    dim3 g2(N_NODES / 128, N_NODES / 128);
    gemm_kernel<<<g2, 256, 0, stream>>>(Xhi, Xlo, sq, R);

    rowstats_kernel<<<N_NODES, 256, 0, stream>>>(R, rowsum, psum, pcnt, nsum, ncnt);
    tau_kernel<<<1, 256, 0, stream>>>(rowsum, psum, pcnt, nsum, ncnt, tau);
    finalize_kernel<<<4096, 256, 0, stream>>>(R, rowsum, tau);
}
